// MarkovTransition_50637664420500
// MI455X (gfx1250) — 8-bit matrix operand order repaired and checked by reading, compiled, not run on hardware
//
#include <hip/hip_runtime.h>

typedef int v8i __attribute__((ext_vector_type(8)));

namespace {
constexpr int NS      = 128;
constexpr int SEQ     = 8192;
constexpr int THREADS = 128;
}

__global__ __launch_bounds__(THREADS)
void markov_transition_wmma(const float* __restrict__ x,
                            float* __restrict__ out)
{
    __shared__ float smin[THREADS];
    __shared__ float smax[THREADS];
    __shared__ unsigned char sidx[SEQ + 16];

    const int tid = threadIdx.x;
    const int bf  = blockIdx.x;
    const int b   = bf >> 3;
    const int f   = bf & 7;
    const float* xs = x + (size_t)b * (SEQ * 8) + f;

    float lmin =  3.402823466e38f;
    float lmax = -3.402823466e38f;
    for (int k = tid; k < SEQ; k += THREADS) {
        float v = xs[(size_t)k * 8];
        lmin = fminf(lmin, v);
        lmax = fmaxf(lmax, v);
    }
    smin[tid] = lmin;
    smax[tid] = lmax;
    __syncthreads();
    for (int s = THREADS / 2; s > 0; s >>= 1) {
        if (tid < s) {
            smin[tid] = fminf(smin[tid], smin[tid + s]);
            smax[tid] = fmaxf(smax[tid], smax[tid + s]);
        }
        __syncthreads();
    }
    const float xmin  = smin[0];
    const float denom = fmaxf(smax[0] - xmin, 1e-8f);

    for (int k = tid; k < SEQ; k += THREADS) {
        float v  = xs[(size_t)k * 8];
        float sc = ((v - xmin) / denom) * 127.0f;
        sc = fminf(fmaxf(sc, 0.0f), 127.0f);
        sidx[k] = (unsigned char)(int)sc;
    }
    if (tid < 16) sidx[SEQ + tid] = 0xFF;
    __syncthreads();

    const int wave = tid >> 5;
    const int lane = tid & 31;
    const int half = lane >> 4;
    const int m    = lane & 15;
    const int row0 = (blockIdx.y * 4 + wave) * 16;
    const unsigned rowpat = (unsigned)(row0 + m) * 0x01010101u;

    v8i acc[8] = {};

    const unsigned* idx32 = (const unsigned*)sidx;

    for (int chunk = 0; chunk < SEQ / 64; ++chunk) {
        const int base = chunk * 64;

        v8i a;
        #pragma unroll
        for (int v = 0; v < 8; ++v) {
            const int koff = 16 * (v >> 1) + 4 * (v & 1) + 8 * half;
            unsigned dw = idx32[(unsigned)(base + koff) >> 2];
            unsigned xv = dw ^ rowpat;
            unsigned t2 = (xv & 0x7f7f7f7fu) + 0x7f7f7f7fu;
            unsigned zm = ~(t2 | xv) & 0x80808080u;
            a[v] = (int)(zm >> 7);
        }

        const unsigned char* sb = sidx + base + 1;

        #pragma unroll
        for (int t = 0; t < 8; ++t) {
            v8i bm;
            const unsigned colpat = (unsigned)(16 * t + m) * 0x01010101u;
            #pragma unroll
            for (int v = 0; v < 8; ++v) {
                const int koff = 16 * (v >> 1) + 4 * (v & 1) + 8 * half;
                unsigned w = (unsigned)sb[koff] | ((unsigned)sb[koff + 1] << 8) |
                             ((unsigned)sb[koff + 2] << 16) | ((unsigned)sb[koff + 3] << 24);
                unsigned xv = w ^ colpat;
                unsigned t2 = (xv & 0x7f7f7f7fu) + 0x7f7f7f7fu;
                unsigned zm = ~(t2 | xv) & 0x80808080u;
                bm[v] = (int)(zm >> 7);
            }
            acc[t] = __builtin_amdgcn_wmma_i32_16x16x64_iu8(
                false, a, false, bm, acc[t], false, false);
        }
    }

    float* outp = out + (size_t)bf * (NS * NS);
    #pragma unroll
    for (int p = 0; p < 8; ++p) {
        int rs = 0;
        #pragma unroll
        for (int t = 0; t < 8; ++t) rs += acc[t][p];
        #pragma unroll
        for (int mk = 1; mk < 16; mk <<= 1) rs += __shfl_xor(rs, mk, 32);
        const float inv = 1.0f / fmaxf((float)rs, 1e-8f);
        const int row = row0 + p + 8 * half;
        #pragma unroll
        for (int t = 0; t < 8; ++t)
            outp[row * NS + 16 * t + m] = (float)acc[t][p] * inv;
    }
}

extern "C" void kernel_launch(void* const* d_in, const int* in_sizes, int n_in,
                              void* d_out, int out_size, void* d_ws, size_t ws_size,
                              hipStream_t stream) {
    (void)in_sizes; (void)n_in; (void)d_ws; (void)ws_size; (void)out_size;
    const float* x = (const float*)d_in[0];
    float* out     = (float*)d_out;
    markov_transition_wmma<<<dim3(32, 2), THREADS, 0, stream>>>(x, out);
}
